// TMSwinBlock_71992241815974
// MI455X (gfx1250) — matrix operand order repaired and checked by reading, not compiled, not run on hardware
//
#include <hip/hip_runtime.h>
#include <hip/hip_bf16.h>
#include <math.h>
#include <stdint.h>

// ---------------- problem constants ----------------
#define BB    64
#define HH    56
#define WW_   56
#define CC    128
#define NHD   4
#define WS    7
#define SHIFT 3
#define HID   512
#define NTOK  49                 // tokens per window
#define NP    64                 // padded tokens per window
#define NWI   64                 // windows per image (8x8)
#define NWIN  (BB * NWI)         // 4096 windows total
#define TOK   (BB * HH * WW_)    // 200704 tokens
#define MQ    (NWIN * NP)        // 262144 padded rows for qkv GEMM

typedef _Float16 h16 __attribute__((ext_vector_type(16)));
typedef _Float16 h8  __attribute__((ext_vector_type(8)));
typedef float    f8  __attribute__((ext_vector_type(8)));
typedef unsigned int u32x4 __attribute__((ext_vector_type(4)));
typedef int          i32x4 __attribute__((ext_vector_type(4)));
typedef int          i32x8 __attribute__((ext_vector_type(8)));

#if defined(__has_builtin)
#  if __has_builtin(__builtin_amdgcn_tensor_load_to_lds)
#    define HAS_TDM 1
#  endif
#endif

__device__ __forceinline__ f8 zero8() {
  f8 z;
#pragma unroll
  for (int i = 0; i < 8; ++i) z[i] = 0.0f;
  return z;
}

// A fragment: 16x32 f16 (M x K). lane<16: row=lane, K kk..kk+15;
// lane>=16: row=lane-16, K kk+16..kk+31.
__device__ __forceinline__ h16 load_a(const _Float16* base, int ld, int mbase, int kk, int lane) {
  int m = mbase + (lane & 15);
  int s = (lane >> 4) * 16;
  return *(const h16*)(base + (size_t)m * ld + kk + s);
}

// B fragment: 32x16 f16 (K x N) from a transposed [N,K] array:
// lane<16: col=lane, K kk..kk+15; lane>=16: col=lane-16, K kk+16..kk+31.
__device__ __forceinline__ h16 load_b(const _Float16* baseT, int ld, int nbase, int kk, int lane) {
  int n = nbase + (lane & 15);
  int s = (lane >> 4) * 16;
  return *(const h16*)(baseT + (size_t)n * ld + kk + s);
}

#define WMMA_F16(a, b, c) \
  __builtin_amdgcn_wmma_f32_16x16x32_f16(false, (a), false, (b), (short)0, (c), false, false)

// ---- TDM: DMA a 2D f16 tile [tl1 rows x tl0 cols] (row stride st0 elements)
// from global into LDS (packed, ld = tl0). D# layout per CDNA5 ISA ch.8. ----
__device__ __forceinline__ void tdm_load_2d(unsigned lds_off, const void* gptr,
                                            unsigned td0, unsigned td1,
                                            unsigned tl0, unsigned tl1, unsigned st0) {
#ifdef HAS_TDM
  unsigned long long ga = (unsigned long long)(uintptr_t)gptr;
  u32x4 g0;
  i32x8 g1;
  i32x4 g2, g3;
#pragma unroll
  for (int i = 0; i < 4; ++i) { g2[i] = 0; g3[i] = 0; }
  g0[0] = 1u;                                            // count=1, user mode
  g0[1] = lds_off;                                       // lds_addr (bytes)
  g0[2] = (unsigned)ga;                                  // global_addr[31:0]
  g0[3] = (unsigned)((ga >> 32) & 0x01FFFFFFu) | 0x80000000u;  // addr[56:32] | type=2
  g1[0] = (int)(1u << 16);                               // data_size = 2 bytes
  g1[1] = (int)((td0 & 0xFFFFu) << 16);                  // tensor_dim0[15:0]
  g1[2] = (int)((td0 >> 16) | ((td1 & 0xFFFFu) << 16));  // dim0 hi | tensor_dim1 lo
  g1[3] = (int)((td1 >> 16) | (tl0 << 16));              // dim1 hi | tile_dim0
  g1[4] = (int)tl1;                                      // tile_dim1 (tile_dim2=0)
  g1[5] = (int)st0;                                      // tensor_dim0_stride lo
  g1[6] = 0;
  g1[7] = 0;
#if __clang_major__ >= 23
  i32x8 g4;
#pragma unroll
  for (int i = 0; i < 8; ++i) g4[i] = 0;
  __builtin_amdgcn_tensor_load_to_lds(g0, g1, g2, g3, g4, 0);
#else
  __builtin_amdgcn_tensor_load_to_lds(g0, g1, g2, g3, 0);
#endif
#endif
}

// ---------------- K0: weight transpose/convert + bias/mask tables ----------------
__global__ void k0_prep(const float* __restrict__ wqkv, const float* __restrict__ wproj,
                        const float* __restrict__ win_, const float* __restrict__ wcore,
                        const float* __restrict__ relb, const int* __restrict__ reli,
                        const float* __restrict__ amask,
                        _Float16* __restrict__ wqkvT, _Float16* __restrict__ wprojT,
                        _Float16* __restrict__ winT, _Float16* __restrict__ wcoreT,
                        float* __restrict__ biaspad, float* __restrict__ maskpad) {
  int i = blockIdx.x * blockDim.x + threadIdx.x;
  if (i < 49152) {               // wqkvT [384,128]
    int n = i / 128, k = i % 128;
    wqkvT[i] = (_Float16)wqkv[k * 384 + n];
    return;
  }
  i -= 49152;
  if (i < 16384) {               // wprojT [128,128]
    int n = i / 128, k = i % 128;
    wprojT[i] = (_Float16)wproj[k * 128 + n];
    return;
  }
  i -= 16384;
  if (i < 131072) {              // winT [1024,128]
    int n = i / 128, k = i % 128;
    winT[i] = (_Float16)win_[k * 1024 + n];
    return;
  }
  i -= 131072;
  if (i < 65536) {               // wcoreT [128,512]
    int n = i / 512, k = i % 512;
    wcoreT[i] = (_Float16)wcore[k * 128 + n];
    return;
  }
  i -= 65536;
  if (i < 16384) {               // biaspad [4][64][64]
    int h = i / 4096, rem = i % 4096, q = rem / 64, kk = rem % 64;
    biaspad[i] = (q < NTOK && kk < NTOK) ? relb[reli[q * NTOK + kk] * NHD + h] : 0.0f;
    return;
  }
  i -= 16384;
  if (i < 262144) {              // maskpad [64][64][64]
    int rem = i % 4096, q = rem / 64, kk = rem % 64, nw = i / 4096;
    maskpad[i] = (kk >= NTOK) ? -1.0e30f : (q < NTOK ? amask[nw * (NTOK * NTOK) + q * NTOK + kk] : 0.0f);
    return;
  }
}

// ---------------- K1: LN1 + cyclic shift + window partition -> f16 ----------------
__global__ void k1_ln_part(const float* __restrict__ x, const float* __restrict__ g1,
                           const float* __restrict__ b1, _Float16* __restrict__ win) {
  int w = blockIdx.x;
  int wv = threadIdx.x >> 5, lane = threadIdx.x & 31;
  int b = w >> 6, wi = w & 63, wr = wi >> 3, wc = wi & 7;
  int c0 = lane * 4;
  for (int t = wv; t < NP; t += 8) {
    _Float16* dst = win + (size_t)w * (NP * CC) + t * CC + c0;
    if (t < NTOK) {
      int hh = wr * WS + t / WS, ww = wc * WS + t % WS;
      int oh = (hh + SHIFT) % HH, ow = (ww + SHIFT) % WW_;
      size_t tok = (size_t)b * (HH * WW_) + oh * WW_ + ow;
      float4 xv = *(const float4*)(x + tok * CC + c0);
      float lsum = xv.x + xv.y + xv.z + xv.w;
#pragma unroll
      for (int off = 1; off < 32; off <<= 1) lsum += __shfl_xor(lsum, off, 32);
      float mean = lsum * (1.0f / 128.0f);
      float d0 = xv.x - mean, d1 = xv.y - mean, d2 = xv.z - mean, d3 = xv.w - mean;
      float lsq = d0 * d0 + d1 * d1 + d2 * d2 + d3 * d3;
#pragma unroll
      for (int off = 1; off < 32; off <<= 1) lsq += __shfl_xor(lsq, off, 32);
      float rs = rsqrtf(lsq * (1.0f / 128.0f) + 1e-5f);
      dst[0] = (_Float16)(d0 * rs * g1[c0 + 0] + b1[c0 + 0]);
      dst[1] = (_Float16)(d1 * rs * g1[c0 + 1] + b1[c0 + 1]);
      dst[2] = (_Float16)(d2 * rs * g1[c0 + 2] + b1[c0 + 2]);
      dst[3] = (_Float16)(d3 * rs * g1[c0 + 3] + b1[c0 + 3]);
    } else {
#pragma unroll
      for (int i = 0; i < 4; ++i) dst[i] = (_Float16)0.0f;
    }
  }
}

// ---------------- K2: qkv GEMM  [MQ,128] x [128,384] -> f16 (pipelined) ----------------
__global__ void k2_qkv(const _Float16* __restrict__ win, const _Float16* __restrict__ wqkvT,
                       const float* __restrict__ bqkv, _Float16* __restrict__ qkv) {
  int wv = threadIdx.x >> 5, lane = threadIdx.x & 31;
  int mtile = wv & 3, nb = (wv >> 2) * 4;
  int mbase = blockIdx.x * 64 + mtile * 16;
  int cb = blockIdx.y * 128;
  f8 acc[4];
#pragma unroll
  for (int j = 0; j < 4; ++j) acc[j] = zero8();
  h16 a_cur = load_a(win, CC, mbase, 0, lane);
  h16 b_cur[4];
#pragma unroll
  for (int j = 0; j < 4; ++j) b_cur[j] = load_b(wqkvT, CC, cb + (nb + j) * 16, 0, lane);
#pragma unroll
  for (int kk = 0; kk < 128; kk += 32) {
    h16 a_nxt = a_cur;
    h16 b_nxt[4];
#pragma unroll
    for (int j = 0; j < 4; ++j) b_nxt[j] = b_cur[j];
    if (kk + 32 < 128) {
      a_nxt = load_a(win, CC, mbase, kk + 32, lane);
#pragma unroll
      for (int j = 0; j < 4; ++j)
        b_nxt[j] = load_b(wqkvT, CC, cb + (nb + j) * 16, kk + 32, lane);
    }
#pragma unroll
    for (int j = 0; j < 4; ++j) acc[j] = WMMA_F16(a_cur, b_cur[j], acc[j]);
    a_cur = a_nxt;
#pragma unroll
    for (int j = 0; j < 4; ++j) b_cur[j] = b_nxt[j];
  }
#pragma unroll
  for (int j = 0; j < 4; ++j) {
    int col = cb + (nb + j) * 16 + (lane & 15);
    float bias = bqkv[col];
#pragma unroll
    for (int r = 0; r < 8; ++r) {
      size_t row = (size_t)mbase + r + 8 * (lane >> 4);
      qkv[row * 384 + col] = (_Float16)(acc[j][r] + bias);
    }
  }
}

// ---------------- K3: windowed attention (one wave per head) ----------------
__global__ void k3_attn(const _Float16* __restrict__ qkv, const float* __restrict__ biaspad,
                        const float* __restrict__ maskpad, _Float16* __restrict__ attn) {
  __shared__ __align__(16) _Float16 vT[NHD][32 * 64];   // [head][hd][key]
  __shared__ __align__(16) _Float16 P[NHD][64 * 64];    // softmax probs
  int w = blockIdx.x;
  int wi = w & 63;
  int hvd = threadIdx.x >> 5;   // head
  int lane = threadIdx.x & 31;
  int halfsel = lane >> 4, cbase = lane & 15;
  const _Float16* base = qkv + (size_t)w * (NP * 384);

  // stage V transposed: vT[d][key] = v[key][d]
  for (int i = lane; i < 32 * 64; i += 32) {
    int d = i >> 6, key = i & 63;
    vT[hvd][d * 64 + key] = base[(size_t)key * 384 + 256 + hvd * 32 + d];
  }

  // S = q k^T  (4x4 tiles of 16x16, K = HD = 32)
  f8 acc[4][4];
  h16 qf[4];
#pragma unroll
  for (int mt = 0; mt < 4; ++mt) qf[mt] = load_a(base + hvd * 32, 384, mt * 16, 0, lane);
#pragma unroll
  for (int nt = 0; nt < 4; ++nt) {
    h16 kf = load_b(base + 128 + hvd * 32, 384, nt * 16, 0, lane);
#pragma unroll
    for (int mt = 0; mt < 4; ++mt) {
      f8 z = zero8();
      acc[mt][nt] = WMMA_F16(qf[mt], kf, z);
    }
  }

  const float scale = 0.17677669529663687f;  // 1/sqrt(32)
  // scale + rel-bias + shift mask, row softmax, write P (f16) to LDS
#pragma unroll
  for (int mt = 0; mt < 4; ++mt) {
#pragma unroll
    for (int r = 0; r < 8; ++r) {
      int row = mt * 16 + r + 8 * halfsel;
      float v0[4];
      float mx = -3.0e38f;
#pragma unroll
      for (int nt = 0; nt < 4; ++nt) {
        int col = nt * 16 + cbase;
        float s = acc[mt][nt][r] * scale + biaspad[hvd * 4096 + row * 64 + col]
                  + maskpad[wi * 4096 + row * 64 + col];
        v0[nt] = s;
        mx = fmaxf(mx, s);
      }
#pragma unroll
      for (int off = 1; off < 16; off <<= 1) mx = fmaxf(mx, __shfl_xor(mx, off, 32));
      float sum = 0.0f;
#pragma unroll
      for (int nt = 0; nt < 4; ++nt) { v0[nt] = expf(v0[nt] - mx); sum += v0[nt]; }
#pragma unroll
      for (int off = 1; off < 16; off <<= 1) sum += __shfl_xor(sum, off, 32);
      float inv = 1.0f / sum;
#pragma unroll
      for (int nt = 0; nt < 4; ++nt)
        P[hvd][row * 64 + nt * 16 + cbase] = (_Float16)(v0[nt] * inv);
    }
  }
  __syncthreads();

  // O = P (64x64) x V (64x32): 4x2 tiles, K loop of 2
#pragma unroll
  for (int mt = 0; mt < 4; ++mt) {
#pragma unroll
    for (int dt = 0; dt < 2; ++dt) {
      f8 o = zero8();
#pragma unroll
      for (int kt = 0; kt < 2; ++kt) {
        h16 a = load_a(&P[hvd][0], 64, mt * 16, kt * 32, lane);
        h16 bf = load_b(&vT[hvd][0], 64, dt * 16, kt * 32, lane);
        o = WMMA_F16(a, bf, o);
      }
      int d = hvd * 32 + dt * 16 + cbase;
#pragma unroll
      for (int r = 0; r < 8; ++r) {
        int row = mt * 16 + r + 8 * halfsel;
        attn[(size_t)w * (NP * CC) + row * CC + d] = (_Float16)o[r];
      }
    }
  }
}

// ---------------- K4: proj GEMM + residual + LN2 (per window) ----------------
__global__ void k4_proj_ln(const _Float16* __restrict__ attn, const _Float16* __restrict__ wprojT,
                           const float* __restrict__ bproj, const float* __restrict__ x,
                           const float* __restrict__ g2, const float* __restrict__ b2,
                           float* __restrict__ x1, _Float16* __restrict__ y2) {
  __shared__ __align__(16) float res[NP * CC];
  int w = blockIdx.x;
  int wv = threadIdx.x >> 5, lane = threadIdx.x & 31;
  int mtile = wv & 3, nb = (wv >> 2) * 4;
  const _Float16* A = attn + (size_t)w * (NP * CC);
  f8 acc[4];
#pragma unroll
  for (int j = 0; j < 4; ++j) acc[j] = zero8();
#pragma unroll
  for (int kk = 0; kk < 128; kk += 32) {
    h16 a = load_a(A, CC, mtile * 16, kk, lane);
#pragma unroll
    for (int j = 0; j < 4; ++j) {
      h16 bf = load_b(wprojT, CC, (nb + j) * 16, kk, lane);
      acc[j] = WMMA_F16(a, bf, acc[j]);
    }
  }
#pragma unroll
  for (int j = 0; j < 4; ++j) {
    int col = (nb + j) * 16 + (lane & 15);
    float bias = bproj[col];
#pragma unroll
    for (int r = 0; r < 8; ++r) {
      int row = mtile * 16 + r + 8 * (lane >> 4);
      res[row * CC + col] = acc[j][r] + bias;
    }
  }
  __syncthreads();

  // window reverse + unshift + residual + LN2
  int b = w >> 6, wi = w & 63, wr = wi >> 3, wc = wi & 7;
  int c0 = lane * 4;
  for (int t = wv; t < NTOK; t += 8) {
    int hh = wr * WS + t / WS, ww = wc * WS + t % WS;
    int oh = (hh + SHIFT) % HH, ow = (ww + SHIFT) % WW_;
    size_t tok = (size_t)b * (HH * WW_) + oh * WW_ + ow;
    float4 xv = *(const float4*)(x + tok * CC + c0);
    float4 rv = *(const float4*)(&res[t * CC + c0]);
    float r0 = xv.x + rv.x, r1 = xv.y + rv.y, r2 = xv.z + rv.z, r3 = xv.w + rv.w;
    float lsum = r0 + r1 + r2 + r3;
#pragma unroll
    for (int off = 1; off < 32; off <<= 1) lsum += __shfl_xor(lsum, off, 32);
    float mean = lsum * (1.0f / 128.0f);
    float d0 = r0 - mean, d1 = r1 - mean, d2 = r2 - mean, d3 = r3 - mean;
    float lsq = d0 * d0 + d1 * d1 + d2 * d2 + d3 * d3;
#pragma unroll
    for (int off = 1; off < 32; off <<= 1) lsq += __shfl_xor(lsq, off, 32);
    float rs = rsqrtf(lsq * (1.0f / 128.0f) + 1e-5f);
    float4 ov;
    ov.x = r0; ov.y = r1; ov.z = r2; ov.w = r3;
    *(float4*)(x1 + tok * CC + c0) = ov;
    _Float16* yd = y2 + tok * CC + c0;
    yd[0] = (_Float16)(d0 * rs * g2[c0 + 0] + b2[c0 + 0]);
    yd[1] = (_Float16)(d1 * rs * g2[c0 + 1] + b2[c0 + 1]);
    yd[2] = (_Float16)(d2 * rs * g2[c0 + 2] + b2[c0 + 2]);
    yd[3] = (_Float16)(d3 * rs * g2[c0 + 3] + b2[c0 + 3]);
  }
}

// ---------------- K6: MLP-in GEMM with fused GEGLU (TDM-staged weights) ----------------
__global__ void k6_mlp_in(const _Float16* __restrict__ y2, const _Float16* __restrict__ winT,
                          const float* __restrict__ bin, _Float16* __restrict__ hbuf) {
  __shared__ __align__(16) _Float16 vlds[64 * CC];   // 16 KB: val weight rows
  __shared__ __align__(16) _Float16 glds[64 * CC];   // 16 KB: gate weight rows
  int wv = threadIdx.x >> 5, lane = threadIdx.x & 31;
  int mtile = wv & 3, nt0 = (wv >> 2) * 2;
  int mbase = blockIdx.x * 64 + mtile * 16;
  int cg = blockIdx.y;  // 0..7 groups of 64 val columns
#ifdef HAS_TDM
  if (wv == 0) {
    tdm_load_2d((unsigned)(uintptr_t)(void*)vlds, winT + (size_t)cg * 64 * CC,
                CC, 64, CC, 64, CC);
    tdm_load_2d((unsigned)(uintptr_t)(void*)glds, winT + (size_t)(HID + cg * 64) * CC,
                CC, 64, CC, 64, CC);
    __builtin_amdgcn_s_wait_tensorcnt(0);
  }
#else
  for (int i = threadIdx.x; i < 64 * 16; i += 256) {
    int n = i >> 4, c = (i & 15) * 8;
    *(h8*)(vlds + n * CC + c) = *(const h8*)(winT + (size_t)(cg * 64 + n) * CC + c);
    *(h8*)(glds + n * CC + c) = *(const h8*)(winT + (size_t)(HID + cg * 64 + n) * CC + c);
  }
#endif
  __syncthreads();

  f8 av[2], ag[2];
#pragma unroll
  for (int j = 0; j < 2; ++j) { av[j] = zero8(); ag[j] = zero8(); }
  h16 a_cur = load_a(y2, CC, mbase, 0, lane);
#pragma unroll
  for (int kk = 0; kk < 128; kk += 32) {
    h16 a_nxt = a_cur;
    if (kk + 32 < 128) a_nxt = load_a(y2, CC, mbase, kk + 32, lane);
#pragma unroll
    for (int j = 0; j < 2; ++j) {
      int nloc = (nt0 + j) * 16;
      h16 bv = load_b(vlds, CC, nloc, kk, lane);
      h16 bg = load_b(glds, CC, nloc, kk, lane);
      av[j] = WMMA_F16(a_cur, bv, av[j]);
      ag[j] = WMMA_F16(a_cur, bg, ag[j]);
    }
    a_cur = a_nxt;
  }
#pragma unroll
  for (int j = 0; j < 2; ++j) {
    int colv = cg * 64 + (nt0 + j) * 16 + (lane & 15);
    float bv = bin[colv], bg = bin[colv + HID];
#pragma unroll
    for (int r = 0; r < 8; ++r) {
      size_t row = (size_t)mbase + r + 8 * (lane >> 4);
      float v = av[j][r] + bv;
      float g = ag[j][r] + bg;
      float ge = 0.5f * v * (1.0f + tanhf(0.7978845608028654f * (v + 0.044715f * v * v * v)));
      float si = 1.0f / (1.0f + expf(-g));
      hbuf[row * HID + colv] = (_Float16)(ge * si);
    }
  }
}

// ---------------- K7: core GEMM K=512 (TDM-staged weight chunks) ----------------
__global__ void k7_core(const _Float16* __restrict__ hbuf, const _Float16* __restrict__ wcoreT,
                        const float* __restrict__ bcore, const float* __restrict__ x1,
                        float* __restrict__ out) {
  __shared__ __align__(16) _Float16 wlds[CC * 128];   // 32 KB: 128 N rows x 128-K chunk
  int wv = threadIdx.x >> 5, lane = threadIdx.x & 31;
  int mtile = wv & 3, nb = (wv >> 2) * 4;
  int mbase = blockIdx.x * 64 + mtile * 16;
  f8 acc[4];
#pragma unroll
  for (int j = 0; j < 4; ++j) acc[j] = zero8();

  for (int chunk = 0; chunk < 4; ++chunk) {
    int k0 = chunk * 128;
#ifdef HAS_TDM
    if (wv == 0) {
      tdm_load_2d((unsigned)(uintptr_t)(void*)wlds, wcoreT + k0,
                  HID, CC, 128, CC, HID);
      __builtin_amdgcn_s_wait_tensorcnt(0);
    }
#else
    for (int i = threadIdx.x; i < CC * 16; i += 256) {
      int n = i >> 4, c = (i & 15) * 8;
      *(h8*)(wlds + n * 128 + c) = *(const h8*)(wcoreT + (size_t)n * HID + k0 + c);
    }
#endif
    __syncthreads();
    if (chunk < 3)
      __builtin_prefetch(hbuf + ((size_t)mbase + (lane & 15)) * HID + k0 + 128, 0, 0);
    h16 a_cur = load_a(hbuf, HID, mbase, k0, lane);
#pragma unroll
    for (int kk = 0; kk < 128; kk += 32) {
      h16 a_nxt = a_cur;
      if (kk + 32 < 128) a_nxt = load_a(hbuf, HID, mbase, k0 + kk + 32, lane);
#pragma unroll
      for (int j = 0; j < 4; ++j) {
        h16 bf = load_b(wlds, 128, (nb + j) * 16, kk, lane);
        acc[j] = WMMA_F16(a_cur, bf, acc[j]);
      }
      a_cur = a_nxt;
    }
    __syncthreads();
  }
#pragma unroll
  for (int j = 0; j < 4; ++j) {
    int col = (nb + j) * 16 + (lane & 15);
    float bias = bcore[col];
#pragma unroll
    for (int r = 0; r < 8; ++r) {
      size_t row = (size_t)mbase + r + 8 * (lane >> 4);
      out[row * CC + col] = acc[j][r] + bias + x1[row * CC + col];
    }
  }
}

// ---------------- host ----------------
extern "C" void kernel_launch(void* const* d_in, const int* in_sizes, int n_in,
                              void* d_out, int out_size, void* d_ws, size_t ws_size,
                              hipStream_t stream) {
  const float* x      = (const float*)d_in[0];
  const float* g1     = (const float*)d_in[1];
  const float* b1     = (const float*)d_in[2];
  const float* w_qkv  = (const float*)d_in[3];
  const float* b_qkv  = (const float*)d_in[4];
  const float* relb   = (const float*)d_in[5];
  const float* w_proj = (const float*)d_in[6];
  const float* b_proj = (const float*)d_in[7];
  const float* g2     = (const float*)d_in[8];
  const float* b2     = (const float*)d_in[9];
  const float* w_in   = (const float*)d_in[10];
  const float* b_in   = (const float*)d_in[11];
  const float* w_core = (const float*)d_in[12];
  const float* b_core = (const float*)d_in[13];
  const float* amask  = (const float*)d_in[14];
  const int*   reli   = (const int*)d_in[15];

  char* ws = (char*)d_ws;
  size_t off = 0;
  auto alloc = [&](size_t bytes) -> void* {
    void* p = (void*)(ws + off);
    off += (bytes + 255) & ~(size_t)255;
    return p;
  };
  _Float16* wqkvT   = (_Float16*)alloc((size_t)384 * 128 * 2);
  _Float16* wprojT  = (_Float16*)alloc((size_t)128 * 128 * 2);
  _Float16* winT    = (_Float16*)alloc((size_t)1024 * 128 * 2);
  _Float16* wcoreT  = (_Float16*)alloc((size_t)128 * 512 * 2);
  float*    biaspad = (float*)alloc((size_t)NHD * 64 * 64 * 4);
  float*    maskpad = (float*)alloc((size_t)NWI * 64 * 64 * 4);
  _Float16* winbuf  = (_Float16*)alloc((size_t)NWIN * NP * CC * 2);
  _Float16* qkvbuf  = (_Float16*)alloc((size_t)NWIN * NP * 384 * 2);
  _Float16* attnbuf = (_Float16*)alloc((size_t)NWIN * NP * CC * 2);
  float*    x1buf   = (float*)alloc((size_t)TOK * CC * 4);
  _Float16* y2buf   = (_Float16*)alloc((size_t)TOK * CC * 2);
  _Float16* hbuf    = (_Float16*)alloc((size_t)TOK * HID * 2);

  k0_prep<<<2112, 256, 0, stream>>>(w_qkv, w_proj, w_in, w_core, relb, reli, amask,
                                    wqkvT, wprojT, winT, wcoreT, biaspad, maskpad);
  k1_ln_part<<<NWIN, 256, 0, stream>>>(x, g1, b1, winbuf);
  k2_qkv<<<dim3(MQ / 64, 3), 256, 0, stream>>>(winbuf, wqkvT, b_qkv, qkvbuf);
  k3_attn<<<NWIN, 128, 0, stream>>>(qkvbuf, biaspad, maskpad, attnbuf);
  k4_proj_ln<<<NWIN, 256, 0, stream>>>(attnbuf, wprojT, b_proj, x, g2, b2, x1buf, y2buf);
  k6_mlp_in<<<dim3(TOK / 64, 8), 256, 0, stream>>>(y2buf, winT, b_in, hbuf);
  k7_core<<<TOK / 64, 256, 0, stream>>>(hbuf, wcoreT, b_core, x1buf, (float*)d_out);
}
